// CausalSelfAttention_26431228740116
// MI455X (gfx1250) — matrix operand order repaired and checked by reading, not compiled, not run on hardware
//
#include <hip/hip_runtime.h>
#include <hip/hip_bf16.h>

#define DEVI __device__ __forceinline__

typedef __bf16 bf16;
typedef __attribute__((ext_vector_type(16))) __bf16 v16bf;
typedef __attribute__((ext_vector_type(8)))  __bf16 bf16x8;
typedef __attribute__((ext_vector_type(4)))  __bf16 bf16x4;
typedef __attribute__((ext_vector_type(8)))  float  v8f;
typedef __attribute__((ext_vector_type(4)))  float  f4;

union FragAB { v16bf bf; bf16x8 u[2]; };

DEVI v8f wmma_bf16(v16bf a, v16bf b, v8f c) {
  return __builtin_amdgcn_wmma_f32_16x16x32_bf16(false, a, false, b, (short)0, c,
                                                 false, false);
}

// ---------------------------------------------------------------------------
// Kernel 1: qkv = x @ w_attn + b_attn, scattered to q/k/v [B,H,T,D] in bf16.
// Block: 256 thr (8 waves). Tile 128(M) x 128(N), K-step 32.
// ---------------------------------------------------------------------------
__global__ __launch_bounds__(256) void qkv_kernel(const float* __restrict__ x,
                                                  const float* __restrict__ w,
                                                  const float* __restrict__ bias,
                                                  bf16* __restrict__ qkv_ws) {
  __shared__ bf16 As[128][48];  // x tile  [m][k] bf16 (pad to 48 for alignment)
  __shared__ bf16 Ws[128][48];  // w tile transposed [n][k] bf16

  const int tid  = threadIdx.x;
  const int wave = tid >> 5, lane = tid & 31;
  const int half = lane >> 4, lidx = lane & 15;
  const int row0 = blockIdx.x * 128;
  const int n0   = blockIdx.y * 128;
  const int wm0  = (wave >> 2) * 64;  // wave sub-tile: 64 rows
  const int wn0  = (wave & 3)  * 32;  //                32 cols

  v8f acc[4][2];
  #pragma unroll
  for (int i = 0; i < 4; ++i)
    #pragma unroll
    for (int j = 0; j < 2; ++j)
      #pragma unroll
      for (int r = 0; r < 8; ++r) acc[i][j][r] = 0.0f;

  for (int kk = 0; kk < 768; kk += 32) {
    // prefetch next K-step tiles (1 global_prefetch_b8 per thread)
    if (kk + 32 < 768) {
      if (tid < 128) {
        __builtin_prefetch(x + (size_t)(row0 + tid) * 768 + kk + 32, 0, 0);
      } else {
        int tt = tid - 128;
        __builtin_prefetch(w + (size_t)(kk + 32 + (tt >> 2)) * 2304 + n0 +
                               (tt & 3) * 32, 0, 0);
      }
    }
    // stage x tile: 128x32 fp32 -> bf16
    #pragma unroll
    for (int p = 0; p < 4; ++p) {
      int r  = p * 32 + (tid >> 3);
      int kq = (tid & 7) * 4;
      f4 xv = *(const f4*)(x + (size_t)(row0 + r) * 768 + kk + kq);
      bf16x4 bq = {(bf16)xv[0], (bf16)xv[1], (bf16)xv[2], (bf16)xv[3]};
      *(bf16x4*)&As[r][kq] = bq;
    }
    // stage w tile transposed: 32(k) x 128(n) -> Ws[n][k]
    #pragma unroll
    for (int p = 0; p < 4; ++p) {
      int kr = p * 8 + (tid >> 5);
      int nq = (tid & 31) * 4;
      f4 wv = *(const f4*)(w + (size_t)(kk + kr) * 2304 + n0 + nq);
      Ws[nq + 0][kr] = (bf16)wv[0];
      Ws[nq + 1][kr] = (bf16)wv[1];
      Ws[nq + 2][kr] = (bf16)wv[2];
      Ws[nq + 3][kr] = (bf16)wv[3];
    }
    __syncthreads();

    FragAB a[4], b[2];
    #pragma unroll
    for (int mi = 0; mi < 4; ++mi) {
      const bf16* rp = &As[wm0 + mi * 16 + lidx][0];
      a[mi].u[0] = *(const bf16x8*)(rp + 8 * half);       // K = 8*half .. +7
      a[mi].u[1] = *(const bf16x8*)(rp + 16 + 8 * half);  // K = 16+8*half .. +7
    }
    #pragma unroll
    for (int ni = 0; ni < 2; ++ni) {
      const bf16* rp = &Ws[wn0 + ni * 16 + lidx][0];
      b[ni].u[0] = *(const bf16x8*)(rp + 8 * half);       // K = 8*half .. +7
      b[ni].u[1] = *(const bf16x8*)(rp + 16 + 8 * half);  // K = 16+8*half .. +7
    }
    #pragma unroll
    for (int mi = 0; mi < 4; ++mi)
      #pragma unroll
      for (int ni = 0; ni < 2; ++ni)
        acc[mi][ni] = wmma_bf16(a[mi].bf, b[ni].bf, acc[mi][ni]);
    __syncthreads();
  }

  // epilogue: + bias, scatter to q/k/v in [B,H,T,D] bf16.
  // which/head/d-base are uniform (128-col tile never straddles the 768 or 64
  // boundaries), so keep the decomposition in scalar registers.
  const size_t QSZ = (size_t)4 * 12 * 2048 * 64;
  const int which = n0 / 768;        // uniform per block
  const int colb  = n0 % 768 + wn0;  // uniform per wave
  bf16* dst = qkv_ws + (size_t)which * QSZ;
  #pragma unroll
  for (int ni = 0; ni < 2; ++ni) {
    int cb   = colb + ni * 16;       // uniform: column base of this fragment
    int head = cb >> 6;              // uniform
    int d    = (cb & 63) + lidx;
    float bv = bias[n0 + wn0 + ni * 16 + lidx];
    #pragma unroll
    for (int mi = 0; mi < 4; ++mi) {
      #pragma unroll
      for (int r = 0; r < 8; ++r) {
        int mg = row0 + wm0 + mi * 16 + 8 * half + r;
        int bb = mg >> 11, tt = mg & 2047;
        dst[(((size_t)(bb * 12 + head)) * 2048 + tt) * 64 + d] =
            (bf16)(acc[mi][ni][r] + bv);
      }
    }
  }
}

// ---------------------------------------------------------------------------
// Kernel 2: flash attention per (b,h). Block: 256 thr, 128 q rows; each wave
// owns 16 q rows. K/V tiles of 64 keys. D = 64.
// ---------------------------------------------------------------------------
__global__ __launch_bounds__(256) void attn_kernel(const bf16* __restrict__ q,
                                                   const bf16* __restrict__ k,
                                                   const bf16* __restrict__ v,
                                                   bf16* __restrict__ attn_out) {
  __shared__ bf16 Qs[128][72];     // [qrow][d]
  __shared__ bf16 Ks[64][72];      // [krow][d]   (B-frag for Q@K^T)
  __shared__ bf16 Vt[64][72];      // [d][krow]   (B-frag for P@V)
  __shared__ bf16 Ps[8][16][72];   // wave-private P tiles 16x64

  const int tid  = threadIdx.x;
  const int wave = tid >> 5, lane = tid & 31;
  const int half = lane >> 4, lidx = lane & 15;
  const int q0   = blockIdx.x * 128;
  const int bh   = blockIdx.y;
  const size_t base = (size_t)bh * 2048 * 64;

  // stage Q tile once
  #pragma unroll
  for (int p = 0; p < 4; ++p) {
    int r  = p * 32 + (tid >> 3);
    int dq = (tid & 7) * 8;
    *(bf16x8*)&Qs[r][dq] = *(const bf16x8*)(q + base + (size_t)(q0 + r) * 64 + dq);
  }
  __syncthreads();

  FragAB aq[2];
  #pragma unroll
  for (int s = 0; s < 2; ++s) {
    const bf16* rp = &Qs[(wave << 4) + lidx][32 * s];
    aq[s].u[0] = *(const bf16x8*)(rp + 8 * half);
    aq[s].u[1] = *(const bf16x8*)(rp + 16 + 8 * half);
  }

  v8f o[4];
  float m_run[8], l_run[8];
  #pragma unroll
  for (int ni = 0; ni < 4; ++ni)
    #pragma unroll
    for (int r = 0; r < 8; ++r) o[ni][r] = 0.0f;
  #pragma unroll
  for (int r = 0; r < 8; ++r) { m_run[r] = -3.0e38f; l_run[r] = 0.0f; }

  const float scale = 0.03608439182435161f;  // 768^-0.5 (per reference)
  const int kend = q0 + 128;
  for (int kt = 0; kt < kend; kt += 64) {
    __syncthreads();
    // prefetch next K/V tile
    if (kt + 64 < kend) {
      if (tid < 64)
        __builtin_prefetch(k + base + (size_t)(kt + 64 + tid) * 64, 0, 0);
      else if (tid < 128)
        __builtin_prefetch(v + base + (size_t)(kt + tid) * 64, 0, 0);  // tid-64+64
    }
    #pragma unroll
    for (int p = 0; p < 2; ++p) {
      int r  = p * 32 + (tid >> 3);
      int dq = (tid & 7) * 8;
      *(bf16x8*)&Ks[r][dq] = *(const bf16x8*)(k + base + (size_t)(kt + r) * 64 + dq);
      bf16x8 vv = *(const bf16x8*)(v + base + (size_t)(kt + r) * 64 + dq);
      #pragma unroll
      for (int j = 0; j < 8; ++j) Vt[dq + j][r] = vv[j];
    }
    __syncthreads();

    // scores = Q @ K^T  (16x64 per wave)
    v8f sc[4];
    #pragma unroll
    for (int ni = 0; ni < 4; ++ni)
      #pragma unroll
      for (int r = 0; r < 8; ++r) sc[ni][r] = 0.0f;
    #pragma unroll
    for (int s = 0; s < 2; ++s) {
      #pragma unroll
      for (int ni = 0; ni < 4; ++ni) {
        FragAB bk;
        const bf16* rp = &Ks[ni * 16 + lidx][32 * s];
        bk.u[0] = *(const bf16x8*)(rp + 8 * half);
        bk.u[1] = *(const bf16x8*)(rp + 16 + 8 * half);
        sc[ni] = wmma_bf16(aq[s].bf, bk.bf, sc[ni]);
      }
    }

    // scale + causal mask
    const int qrow_base = q0 + (wave << 4) + (half << 3);
    #pragma unroll
    for (int ni = 0; ni < 4; ++ni) {
      int kcol = kt + ni * 16 + lidx;
      #pragma unroll
      for (int r = 0; r < 8; ++r) {
        float sv = sc[ni][r] * scale;
        sc[ni][r] = (kcol > qrow_base + r) ? -3.0e38f : sv;
      }
    }

    // online softmax (row stats across the 16-lane n-groups)
    #pragma unroll
    for (int r = 0; r < 8; ++r) {
      float mx = fmaxf(fmaxf(sc[0][r], sc[1][r]), fmaxf(sc[2][r], sc[3][r]));
      #pragma unroll
      for (int off = 1; off < 16; off <<= 1) mx = fmaxf(mx, __shfl_xor(mx, off));
      float m_new = fmaxf(m_run[r], mx);
      float alpha = __expf(m_run[r] - m_new);
      m_run[r] = m_new;
      float rs = 0.0f;
      #pragma unroll
      for (int ni = 0; ni < 4; ++ni) {
        float pv = __expf(sc[ni][r] - m_new);
        sc[ni][r] = pv;
        rs += pv;
      }
      #pragma unroll
      for (int off = 1; off < 16; off <<= 1) rs += __shfl_xor(rs, off);
      l_run[r] = l_run[r] * alpha + rs;
      #pragma unroll
      for (int ni = 0; ni < 4; ++ni) o[ni][r] *= alpha;
    }

    // D-layout -> A-layout via wave-private LDS round trip
    #pragma unroll
    for (int ni = 0; ni < 4; ++ni)
      #pragma unroll
      for (int r = 0; r < 8; ++r)
        Ps[wave][(half << 3) + r][ni * 16 + lidx] = (bf16)sc[ni][r];
    asm volatile("s_wait_dscnt 0" ::: "memory");

    // o += P @ V
    #pragma unroll
    for (int s = 0; s < 2; ++s) {
      FragAB ap;
      const bf16* rp = &Ps[wave][lidx][32 * s];
      ap.u[0] = *(const bf16x8*)(rp + 8 * half);
      ap.u[1] = *(const bf16x8*)(rp + 16 + 8 * half);
      #pragma unroll
      for (int ni = 0; ni < 4; ++ni) {
        FragAB bv;
        const bf16* vp = &Vt[ni * 16 + lidx][32 * s];
        bv.u[0] = *(const bf16x8*)(vp + 8 * half);
        bv.u[1] = *(const bf16x8*)(vp + 16 + 8 * half);
        o[ni] = wmma_bf16(ap.bf, bv.bf, o[ni]);
      }
    }
  }

  // epilogue: normalize, write [B,T,C] bf16
  const int b = bh / 12, h = bh % 12;
  #pragma unroll
  for (int ni = 0; ni < 4; ++ni) {
    int d = ni * 16 + lidx;
    #pragma unroll
    for (int r = 0; r < 8; ++r) {
      int t = q0 + (wave << 4) + (half << 3) + r;
      attn_out[((size_t)(b * 2048 + t)) * 768 + h * 64 + d] =
          (bf16)(o[ni][r] / l_run[r]);
    }
  }
}

// ---------------------------------------------------------------------------
// Kernel 3: out = attn @ w_proj + b_proj (fp32 out). Same tiling as kernel 1.
// ---------------------------------------------------------------------------
__global__ __launch_bounds__(256) void proj_kernel(const bf16* __restrict__ a_bf,
                                                   const float* __restrict__ w,
                                                   const float* __restrict__ bias,
                                                   float* __restrict__ out) {
  __shared__ bf16 As[128][48];
  __shared__ bf16 Ws[128][48];

  const int tid  = threadIdx.x;
  const int wave = tid >> 5, lane = tid & 31;
  const int half = lane >> 4, lidx = lane & 15;
  const int row0 = blockIdx.x * 128;
  const int n0   = blockIdx.y * 128;
  const int wm0  = (wave >> 2) * 64;
  const int wn0  = (wave & 3)  * 32;

  v8f acc[4][2];
  #pragma unroll
  for (int i = 0; i < 4; ++i)
    #pragma unroll
    for (int j = 0; j < 2; ++j)
      #pragma unroll
      for (int r = 0; r < 8; ++r) acc[i][j][r] = 0.0f;

  for (int kk = 0; kk < 768; kk += 32) {
    if (kk + 32 < 768) {
      if (tid < 128) {
        __builtin_prefetch(a_bf + (size_t)(row0 + tid) * 768 + kk + 32, 0, 0);
      } else {
        int tt = tid - 128;
        __builtin_prefetch(w + (size_t)(kk + 32 + (tt >> 2)) * 768 + n0 +
                               (tt & 3) * 32, 0, 0);
      }
    }
    #pragma unroll
    for (int p = 0; p < 4; ++p) {
      int r  = p * 32 + (tid >> 3);
      int kq = (tid & 7) * 4;
      *(bf16x4*)&As[r][kq] =
          *(const bf16x4*)(a_bf + (size_t)(row0 + r) * 768 + kk + kq);
    }
    #pragma unroll
    for (int p = 0; p < 4; ++p) {
      int kr = p * 8 + (tid >> 5);
      int nq = (tid & 31) * 4;
      f4 wv = *(const f4*)(w + (size_t)(kk + kr) * 768 + n0 + nq);
      Ws[nq + 0][kr] = (bf16)wv[0];
      Ws[nq + 1][kr] = (bf16)wv[1];
      Ws[nq + 2][kr] = (bf16)wv[2];
      Ws[nq + 3][kr] = (bf16)wv[3];
    }
    __syncthreads();

    FragAB a[4], b[2];
    #pragma unroll
    for (int mi = 0; mi < 4; ++mi) {
      const bf16* rp = &As[wm0 + mi * 16 + lidx][0];
      a[mi].u[0] = *(const bf16x8*)(rp + 8 * half);
      a[mi].u[1] = *(const bf16x8*)(rp + 16 + 8 * half);
    }
    #pragma unroll
    for (int ni = 0; ni < 2; ++ni) {
      const bf16* rp = &Ws[wn0 + ni * 16 + lidx][0];
      b[ni].u[0] = *(const bf16x8*)(rp + 8 * half);
      b[ni].u[1] = *(const bf16x8*)(rp + 16 + 8 * half);
    }
    #pragma unroll
    for (int mi = 0; mi < 4; ++mi)
      #pragma unroll
      for (int ni = 0; ni < 2; ++ni)
        acc[mi][ni] = wmma_bf16(a[mi].bf, b[ni].bf, acc[mi][ni]);
    __syncthreads();
  }

  #pragma unroll
  for (int ni = 0; ni < 2; ++ni) {
    int ng   = n0 + wn0 + ni * 16 + lidx;
    float bv = bias[ng];
    #pragma unroll
    for (int mi = 0; mi < 4; ++mi) {
      #pragma unroll
      for (int r = 0; r < 8; ++r) {
        int mg = row0 + wm0 + mi * 16 + 8 * half + r;
        out[(size_t)mg * 768 + ng] = acc[mi][ni][r] + bv;
      }
    }
  }
}

// ---------------------------------------------------------------------------
extern "C" void kernel_launch(void* const* d_in, const int* in_sizes, int n_in,
                              void* d_out, int out_size, void* d_ws, size_t ws_size,
                              hipStream_t stream) {
  (void)in_sizes; (void)n_in; (void)out_size; (void)ws_size;
  const float* x      = (const float*)d_in[0];
  const float* w_attn = (const float*)d_in[1];
  const float* b_attn = (const float*)d_in[2];
  const float* w_proj = (const float*)d_in[3];
  const float* b_proj = (const float*)d_in[4];

  bf16* ws = (bf16*)d_ws;
  const size_t QSZ = (size_t)4 * 12 * 2048 * 64;  // 6,291,456 elems per tensor
  bf16* qkv  = ws;            // q | k | v : 3 * QSZ bf16
  bf16* attn = ws + 3 * QSZ;  // [B,T,C] bf16

  qkv_kernel <<<dim3(64, 18), 256, 0, stream>>>(x, w_attn, b_attn, qkv);
  attn_kernel<<<dim3(16, 48), 256, 0, stream>>>(qkv, qkv + QSZ, qkv + 2 * QSZ, attn);
  proj_kernel<<<dim3(64, 6),  256, 0, stream>>>(attn, w_proj, b_proj, (float*)d_out);
}
